// GCNEncoder_61710090109081
// MI455X (gfx1250) — matrix operand order repaired and checked by reading, not compiled, not run on hardware
//
#include <hip/hip_runtime.h>

// ---------------------------------------------------------------------------
// GCN encoder (VGAE) for MI455X / gfx1250, wave32 + WMMA bf16.
//   N=50000 nodes, E=800000 edges, IN=2000, HID=256, LAT=64
// Workspace layout (floats): deg[N] | dinv[N] | H1[N*256] (reused as Z[N*128]) | AGG[N*256]
//   total ~103 MB.
// ---------------------------------------------------------------------------

constexpr int GN_N   = 50000;
constexpr int GN_E   = 800000;
constexpr int GN_IN  = 2000;
constexpr int GN_H   = 256;
constexpr int GN_L   = 64;
constexpr int GN_EN  = GN_E + GN_N;          // edges + self loops
constexpr int GN_KMAIN = (GN_IN / 32) * 32;  // 1984: unguarded K steps

typedef __attribute__((ext_vector_type(16))) __bf16 v16bf;
typedef __attribute__((ext_vector_type(8)))  __bf16 v8bf;
typedef __attribute__((ext_vector_type(8)))  float  v8f;

// ---------------------------------------------------------------------------
// helpers
// ---------------------------------------------------------------------------
static __device__ __forceinline__ float4 relu4(float4 q) {
  return make_float4(fmaxf(q.x, 0.f), fmaxf(q.y, 0.f),
                     fmaxf(q.z, 0.f), fmaxf(q.w, 0.f));
}

// pack 16 f32 into the ISA 16-bit A-fragment order (native v_cvt RNE)
static __device__ __forceinline__ v16bf pack_a(float4 q0, float4 q1,
                                               float4 q2, float4 q3) {
  v16bf a;
  a[0]  = (__bf16)q0.x; a[1]  = (__bf16)q0.y; a[2]  = (__bf16)q0.z; a[3]  = (__bf16)q0.w;
  a[4]  = (__bf16)q1.x; a[5]  = (__bf16)q1.y; a[6]  = (__bf16)q1.z; a[7]  = (__bf16)q1.w;
  a[8]  = (__bf16)q2.x; a[9]  = (__bf16)q2.y; a[10] = (__bf16)q2.z; a[11] = (__bf16)q2.w;
  a[12] = (__bf16)q3.x; a[13] = (__bf16)q3.y; a[14] = (__bf16)q3.z; a[15] = (__bf16)q3.w;
  return a;
}

static __device__ __forceinline__ v16bf load_bfrag(const __bf16* bp) {
  v8bf lo = *(const v8bf*)bp;
  v8bf hi = *(const v8bf*)(bp + 16);
  return __builtin_shufflevector(lo, hi, 0, 1, 2, 3, 4, 5, 6, 7,
                                 8, 9, 10, 11, 12, 13, 14, 15);
}

// 4 B-fragment ds_load_b128 pairs first, then 4 back-to-back WMMAs
static __device__ __forceinline__ void wmma_step(const __bf16* ldsBk, int ldb,
                                                 const v16bf& a, int l15, int hi,
                                                 v8f acc[4]) {
  const int ko = hi * 8;
  v16bf b0 = load_bfrag(&ldsBk[(0 * 16 + l15) * ldb + ko]);
  v16bf b1 = load_bfrag(&ldsBk[(1 * 16 + l15) * ldb + ko]);
  v16bf b2 = load_bfrag(&ldsBk[(2 * 16 + l15) * ldb + ko]);
  v16bf b3 = load_bfrag(&ldsBk[(3 * 16 + l15) * ldb + ko]);
  acc[0] = __builtin_amdgcn_wmma_f32_16x16x32_bf16(false, a, false, b0, (short)0, acc[0], false, false);
  acc[1] = __builtin_amdgcn_wmma_f32_16x16x32_bf16(false, a, false, b1, (short)0, acc[1], false, false);
  acc[2] = __builtin_amdgcn_wmma_f32_16x16x32_bf16(false, a, false, b2, (short)0, acc[2], false, false);
  acc[3] = __builtin_amdgcn_wmma_f32_16x16x32_bf16(false, a, false, b3, (short)0, acc[3], false, false);
}

// ---------------------------------------------------------------------------
// degree / normalization
// ---------------------------------------------------------------------------
__global__ void gcn_init_deg(float* __restrict__ deg) {
  int i = blockIdx.x * blockDim.x + threadIdx.x;
  if (i < GN_N) deg[i] = 1.0f;                 // self-loop contribution
}

__global__ void gcn_accum_deg(const long long* __restrict__ ei, float* __restrict__ deg) {
  int e = blockIdx.x * blockDim.x + threadIdx.x;
  if (e < GN_E) atomicAdd(&deg[(int)ei[GN_E + e]], 1.0f);
}

__global__ void gcn_dinv(const float* __restrict__ deg, float* __restrict__ dinv) {
  int i = blockIdx.x * blockDim.x + threadIdx.x;
  if (i < GN_N) dinv[i] = rsqrtf(deg[i]);      // deg >= 1 always
}

// ---------------------------------------------------------------------------
// GEMM1: H1[N,256] = X[N,2000] @ W1[2000,256]   (bf16 WMMA, f32 accum)
// block: 256 thr = 8 waves; tile 128(M) x 64(N); wave tile 16 x 64.
// K split: 62 unguarded steps + 1 guarded tail (2000 = 62*32 + 16).
// ---------------------------------------------------------------------------
__global__ __launch_bounds__(256) void gcn_gemm1_wmma(
    const float* __restrict__ X, const float* __restrict__ W,
    float* __restrict__ H) {
  __shared__ __align__(16) __bf16 ldsB[64 * 40];   // [n][k] tile, k padded 32->40

  const int tid  = threadIdx.x;
  const int lane = tid & 31;
  const int wave = tid >> 5;
  const int l15  = lane & 15;
  const int hi   = lane >> 4;                      // half-wave select
  const int mBase   = blockIdx.x * 128 + wave * 16;
  const int colBase = blockIdx.y * 64;

  v8f acc[4];
#pragma unroll
  for (int f = 0; f < 4; ++f)
#pragma unroll
    for (int i = 0; i < 8; ++i) acc[f][i] = 0.0f;

  int rowA = mBase + l15;
  if (rowA >= GN_N) rowA = GN_N - 1;               // clamp loads, stores guarded
  const float* arow = X + (size_t)rowA * GN_IN;

  // ---- main K loop: no bounds checks anywhere -----------------------------
  for (int kk = 0; kk < GN_KMAIN; kk += 32) {
    __syncthreads();
    // stage B tile [32k x 64n]: float4 loads along n, bf16 stores [n][40]
#pragma unroll
    for (int it = 0; it < 2; ++it) {
      const int vid = tid + it * 256;              // 0..511
      const int kl  = vid >> 4;                    // 0..31
      const int n4  = (vid & 15) * 4;              // 0,4,..,60
      const float4 q = *(const float4*)&W[(size_t)(kk + kl) * GN_H + colBase + n4];
      ldsB[(n4 + 0) * 40 + kl] = (__bf16)q.x;
      ldsB[(n4 + 1) * 40 + kl] = (__bf16)q.y;
      ldsB[(n4 + 2) * 40 + kl] = (__bf16)q.z;
      ldsB[(n4 + 3) * 40 + kl] = (__bf16)q.w;
    }
    __syncthreads();

    __builtin_prefetch(arow + kk + 32, 0, 3);

    // A fragment per ISA 16-bit 16x32 layout:
    // lanes 0-15: K {0..7,16..23}; lanes 16-31: K {8..15,24..31}
    const float4* p0 = (const float4*)(arow + kk + hi * 8);
    const float4* p1 = (const float4*)(arow + kk + 16 + hi * 8);
    const v16bf a = pack_a(p0[0], p0[1], p1[0], p1[1]);

    wmma_step(ldsB, 40, a, l15, hi, acc);
  }

  // ---- K tail: kk = 1984, valid K 1984..1999 ------------------------------
  {
    const int kk = GN_KMAIN;
    __syncthreads();
#pragma unroll
    for (int it = 0; it < 2; ++it) {
      const int vid = tid + it * 256;
      const int kl  = vid >> 4;
      const int n4  = (vid & 15) * 4;
      float4 q = make_float4(0.f, 0.f, 0.f, 0.f);
      if (kk + kl < GN_IN)
        q = *(const float4*)&W[(size_t)(kk + kl) * GN_H + colBase + n4];
      ldsB[(n4 + 0) * 40 + kl] = (__bf16)q.x;
      ldsB[(n4 + 1) * 40 + kl] = (__bf16)q.y;
      ldsB[(n4 + 2) * 40 + kl] = (__bf16)q.z;
      ldsB[(n4 + 3) * 40 + kl] = (__bf16)q.w;
    }
    __syncthreads();

    const float4* p0 = (const float4*)(arow + kk + hi * 8);  // K 1984..1999: valid
    const float4 zero = make_float4(0.f, 0.f, 0.f, 0.f);
    const v16bf a = pack_a(p0[0], p0[1], zero, zero);        // K >= 2000 zero-filled

    wmma_step(ldsB, 40, a, l15, hi, acc);
  }

  // C/D layout: lane = col (0-15), VGPR r -> row r (+8 for upper half-wave)
#pragma unroll
  for (int f = 0; f < 4; ++f) {
    const int colg = colBase + f * 16 + l15;
#pragma unroll
    for (int r = 0; r < 8; ++r) {
      const int m = mBase + r + hi * 8;
      if (m < GN_N) H[(size_t)m * GN_H + colg] = acc[f][r];
    }
  }
}

// ---------------------------------------------------------------------------
// AGG prefill with bias (agg + b comes free), then edge scatter-add
// ---------------------------------------------------------------------------
__global__ void gcn_prefill_agg(const float* __restrict__ b1, float* __restrict__ AGG) {
  AGG[(size_t)blockIdx.x * GN_H + threadIdx.x] = b1[threadIdx.x];
}

__global__ void gcn_aggregate_h(const long long* __restrict__ ei,
                                const float* __restrict__ H1,
                                const float* __restrict__ dinv,
                                float* __restrict__ AGG) {
  const int gw   = (blockIdx.x * blockDim.x + threadIdx.x) >> 5;  // one wave = one edge
  const int lane = threadIdx.x & 31;
  if (gw >= GN_EN) return;
  int src, dst;
  if (gw < GN_E) { src = (int)ei[gw]; dst = (int)ei[GN_E + gw]; }
  else           { src = gw - GN_E;   dst = src; }                // self loop
  const float nrm = dinv[src] * dinv[dst];
  const float4* hp = (const float4*)(H1 + (size_t)src * GN_H);
  const float4 a = hp[lane];
  const float4 b = hp[lane + 32];
  float* op = AGG + (size_t)dst * GN_H;
  atomicAdd(op + 4 * lane + 0, a.x * nrm);
  atomicAdd(op + 4 * lane + 1, a.y * nrm);
  atomicAdd(op + 4 * lane + 2, a.z * nrm);
  atomicAdd(op + 4 * lane + 3, a.w * nrm);
  atomicAdd(op + 128 + 4 * lane + 0, b.x * nrm);
  atomicAdd(op + 128 + 4 * lane + 1, b.y * nrm);
  atomicAdd(op + 128 + 4 * lane + 2, b.z * nrm);
  atomicAdd(op + 128 + 4 * lane + 3, b.w * nrm);
}

// ---------------------------------------------------------------------------
// GEMM2: Z[N,128] = relu(AGG)[N,256] @ [W_mu | W_var][256,128]
// Full 64-col bf16 B panel in LDS (33.8 KB); panel source uniform per block.y.
// ReLU fused into A load; K=256 divides 32, no guards at all.
// ---------------------------------------------------------------------------
__global__ __launch_bounds__(256) void gcn_gemm2_wmma(
    const float* __restrict__ A, const float* __restrict__ Wmu,
    const float* __restrict__ Wvar, float* __restrict__ Z) {
  __shared__ __align__(16) __bf16 ldsB[64 * 264];  // [n][k], k padded 256->264

  const int tid  = threadIdx.x;
  const int lane = tid & 31;
  const int wave = tid >> 5;
  const int l15  = lane & 15;
  const int hi   = lane >> 4;
  const int mBase   = blockIdx.x * 128 + wave * 16;
  const int colBase = blockIdx.y * 64;             // 0 -> mu cols, 64 -> var cols

  // whole B panel: source matrix is uniform per block (mu or var)
  const float* __restrict__ Wsel = (blockIdx.y == 0) ? Wmu : Wvar;
#pragma unroll
  for (int it = 0; it < 16; ++it) {
    const int vid = tid + it * 256;                // 0..4095
    const int k   = vid >> 4;                      // 0..255
    const int n4  = (vid & 15) * 4;                // 0..60
    const float4 q = *(const float4*)&Wsel[(size_t)k * GN_L + n4];
    ldsB[(n4 + 0) * 264 + k] = (__bf16)q.x;
    ldsB[(n4 + 1) * 264 + k] = (__bf16)q.y;
    ldsB[(n4 + 2) * 264 + k] = (__bf16)q.z;
    ldsB[(n4 + 3) * 264 + k] = (__bf16)q.w;
  }
  __syncthreads();

  v8f acc[4];
#pragma unroll
  for (int f = 0; f < 4; ++f)
#pragma unroll
    for (int i = 0; i < 8; ++i) acc[f][i] = 0.0f;

  int rowA = mBase + l15;
  if (rowA >= GN_N) rowA = GN_N - 1;
  const float* arow = A + (size_t)rowA * GN_H;

#pragma unroll
  for (int kk = 0; kk < GN_H; kk += 32) {
    const float4* p0 = (const float4*)(arow + kk + hi * 8);
    const float4* p1 = (const float4*)(arow + kk + 16 + hi * 8);
    const v16bf a = pack_a(relu4(p0[0]), relu4(p0[1]), relu4(p1[0]), relu4(p1[1]));
    wmma_step(ldsB + kk, 264, a, l15, hi, acc);
  }

#pragma unroll
  for (int f = 0; f < 4; ++f) {
    const int colg = colBase + f * 16 + l15;
#pragma unroll
    for (int r = 0; r < 8; ++r) {
      const int m = mBase + r + hi * 8;
      if (m < GN_N) Z[(size_t)m * 128 + colg] = acc[f][r];
    }
  }
}

// ---------------------------------------------------------------------------
// output: prefill biases, then edge scatter-add into (mu | log_var)
// ---------------------------------------------------------------------------
__global__ void gcn_prefill_out(const float* __restrict__ bmu,
                                const float* __restrict__ bvar,
                                float* __restrict__ out) {
  const long long i = (long long)blockIdx.x * blockDim.x + threadIdx.x;
  const long long half = (long long)GN_N * GN_L;
  if (i < half)            out[i] = bmu[(int)(i & 63)];
  else if (i < 2 * half)   out[i] = bvar[(int)(i & 63)];
}

__global__ void gcn_aggregate_out(const long long* __restrict__ ei,
                                  const float* __restrict__ Z,
                                  const float* __restrict__ dinv,
                                  float* __restrict__ out) {
  const int gw   = (blockIdx.x * blockDim.x + threadIdx.x) >> 5;
  const int lane = threadIdx.x & 31;
  if (gw >= GN_EN) return;
  int src, dst;
  if (gw < GN_E) { src = (int)ei[gw]; dst = (int)ei[GN_E + gw]; }
  else           { src = gw - GN_E;   dst = src; }
  const float nrm = dinv[src] * dinv[dst];
  const float4 v = ((const float4*)(Z + (size_t)src * 128))[lane];
  const int c0 = 4 * lane;                        // 0..124, never straddles 64
  float* base = (c0 < GN_L)
      ? (out + (size_t)dst * GN_L + c0)                                // mu half
      : (out + (size_t)GN_N * GN_L + (size_t)dst * GN_L + (c0 - GN_L)); // log_var half
  atomicAdd(base + 0, v.x * nrm);
  atomicAdd(base + 1, v.y * nrm);
  atomicAdd(base + 2, v.z * nrm);
  atomicAdd(base + 3, v.w * nrm);
}

// ---------------------------------------------------------------------------
extern "C" void kernel_launch(void* const* d_in, const int* in_sizes, int n_in,
                              void* d_out, int out_size, void* d_ws, size_t ws_size,
                              hipStream_t stream) {
  (void)in_sizes; (void)n_in; (void)out_size; (void)ws_size;

  const float*     x    = (const float*)d_in[0];
  const long long* ei   = (const long long*)d_in[1];   // int64 edge_index [2,E]
  const float*     W1   = (const float*)d_in[2];
  const float*     b1   = (const float*)d_in[3];
  const float*     Wmu  = (const float*)d_in[4];
  const float*     bmu  = (const float*)d_in[5];
  const float*     Wvar = (const float*)d_in[6];
  const float*     bvar = (const float*)d_in[7];
  float* out = (float*)d_out;
  float* ws  = (float*)d_ws;

  float* deg  = ws;                                // N
  float* dinv = ws + GN_N;                         // N
  float* H1   = ws + 2 * GN_N;                     // N*256 (reused as Z: N*128)
  float* AGG  = H1 + (size_t)GN_N * GN_H;          // N*256
  float* Z    = H1;

  const int mblocks = (GN_N + 127) / 128;          // 391
  const int edgeBlocks = ((GN_EN * 32) + 255) / 256;

  gcn_init_deg <<<(GN_N + 255) / 256, 256, 0, stream>>>(deg);
  gcn_accum_deg<<<(GN_E + 255) / 256, 256, 0, stream>>>(ei, deg);
  gcn_dinv     <<<(GN_N + 255) / 256, 256, 0, stream>>>(deg, dinv);

  gcn_gemm1_wmma<<<dim3(mblocks, GN_H / 64), 256, 0, stream>>>(x, W1, H1);

  gcn_prefill_agg<<<GN_N, GN_H, 0, stream>>>(b1, AGG);
  gcn_aggregate_h<<<edgeBlocks, 256, 0, stream>>>(ei, H1, dinv, AGG);

  gcn_gemm2_wmma<<<dim3(mblocks, 2), 256, 0, stream>>>(AGG, Wmu, Wvar, Z);

  gcn_prefill_out<<<(2 * GN_N * GN_L + 255) / 256, 256, 0, stream>>>(bmu, bvar, out);
  gcn_aggregate_out<<<edgeBlocks, 256, 0, stream>>>(ei, Z, dinv, out);
}
